// HypModel_57277683859539
// MI455X (gfx1250) — matrix operand order repaired and checked by reading, not compiled, not run on hardware
//
#include <hip/hip_runtime.h>
#include <hip/hip_bf16.h>

// ---------------------------------------------------------------------------
// MI455X (gfx1250, wave32) implementation of the stock-GNN reference.
//  - GEMMs run on v_wmma_f32_16x16x32_bf16 (bf16 A/B, f32 accumulate)
//  - each wave computes a 16(M) x 64(N) tile: one A fragment feeds 4 WMMAs,
//    cutting VMEM traffic from 4 -> 2.5 b128 loads per WMMA
//  - activations stored bf16 (halves HBM traffic; workload is BW-bound:
//    ~38 GFLOP vs ~300MB of traffic at 23.3 TB/s)
//  - graph scatter-add uses native f32 global atomics (L2-resident: y = 64MB
//    fits in the 192MB L2)
// ---------------------------------------------------------------------------

typedef __bf16 bf16;
typedef __attribute__((ext_vector_type(16))) __bf16 v16bf;
typedef __attribute__((ext_vector_type(8)))  __bf16 v8bf;
typedef __attribute__((ext_vector_type(8)))  float  v8f;

#define N_STOCKS 4096
#define NUM_DAYS 8
#define IN_DIM   64
#define HID      512
#define HID2     256
#define N_EDGES  (N_STOCKS * 64)
#define MROWS    (NUM_DAYS * N_STOCKS)   // 32768

__device__ __forceinline__ float lrelu(float v) { return v > 0.f ? v : 0.01f * v; }

// ---------------- elementwise helpers ----------------

__global__ void k_f32_to_bf16(const float* __restrict__ in, bf16* __restrict__ out, int n) {
    int i = blockIdx.x * blockDim.x + threadIdx.x;
    if (i < n) out[i] = (bf16)in[i];
}

__global__ void k_fill_f32(float* __restrict__ p, float v, int n) {
    int i = blockIdx.x * blockDim.x + threadIdx.x;
    if (i < n) p[i] = v;
}

// ---------------- WMMA GEMM ----------------
// C[M,N] = lrelu?(A[M,K](bf16) * B[N,K]^T(bf16) + bias[N]); out f32 or bf16.
// One wave -> 16(M) x 64(N) tile (4 accumulators share one A fragment).
// Block = 4 waves -> 16 x 256 of N per block; grid = (N/256, M/16).
//
// VGPR layouts used here (A/B in contiguous k order, not the ISA table order; valid since A and B match):
//  A (16x32 bf16): lane L holds row M=L&15; K values khalf..khalf+15 with
//      khalf = (L>=16)?16:0                        -> one 32B contiguous load
//  B (32x16 bf16): lane L holds col N=L&15; K values khalf..khalf+15 with
//      khalf = (L>=16)?16:0                        -> one 32B contiguous load
//  C/D (16x16 f32): c[r] = D[M = r + 8*(L>=16)][N = L&15]
template<bool LRELU, bool OUT_BF16>
__global__ void k_gemm_wmma(const bf16* __restrict__ A, const bf16* __restrict__ B,
                            const float* __restrict__ bias, void* __restrict__ Out,
                            int M, int N, int K)
{
    const int lane = threadIdx.x & 31;
    const int wave = threadIdx.x >> 5;
    const int n0 = (blockIdx.x * 4 + wave) * 64;   // wave's first output column
    const int m0 = blockIdx.y * 16;
    if (n0 >= N) return;                 // wave-uniform exit; EXEC stays all-1s

    const int hh  = lane >> 4;           // 0 or 1 (lane half)
    const int idx = lane & 15;

    const bf16* Arow = A + (size_t)(m0 + idx) * K;
    const bf16* Brow = B + (size_t)(n0 + idx) * K;   // col-tile t: + t*16*K

    v8f c0 = {0.f,0.f,0.f,0.f,0.f,0.f,0.f,0.f};
    v8f c1 = c0, c2 = c0, c3 = c0;

    for (int k0 = 0; k0 < K; k0 += 32) {
        v16bf a = *(const v16bf*)(Arow + k0 + hh * 16);

        v16bf b0 = *(const v16bf*)(Brow + (size_t)0 * 16 * K + k0 + hh * 16);
        v16bf b1 = *(const v16bf*)(Brow + (size_t)1 * 16 * K + k0 + hh * 16);
        v16bf b2 = *(const v16bf*)(Brow + (size_t)2 * 16 * K + k0 + hh * 16);
        v16bf b3 = *(const v16bf*)(Brow + (size_t)3 * 16 * K + k0 + hh * 16);

        // (neg_a, A, neg_b, B, c_mod, C, reuse_a, reuse_b)
        c0 = __builtin_amdgcn_wmma_f32_16x16x32_bf16(false, a, false, b0, (short)0, c0, false, false);
        c1 = __builtin_amdgcn_wmma_f32_16x16x32_bf16(false, a, false, b1, (short)0, c1, false, false);
        c2 = __builtin_amdgcn_wmma_f32_16x16x32_bf16(false, a, false, b2, (short)0, c2, false, false);
        c3 = __builtin_amdgcn_wmma_f32_16x16x32_bf16(false, a, false, b3, (short)0, c3, false, false);
    }

    v8f acc[4] = {c0, c1, c2, c3};
#pragma unroll
    for (int t = 0; t < 4; ++t) {
        const int   nc = n0 + t * 16 + idx;
        const float bv = bias[nc];
#pragma unroll
        for (int r = 0; r < 8; ++r) {
            float v = acc[t][r] + bv;
            if (LRELU) v = lrelu(v);
            const size_t o = (size_t)(m0 + r + hh * 8) * N + nc;
            if constexpr (OUT_BF16) ((bf16*)Out)[o] = (bf16)v;
            else                    ((float*)Out)[o] = v;
        }
    }
}

// ---------------- graph aggregation ----------------

__global__ void k_deg(const int* __restrict__ dst, float* __restrict__ deg, int nE) {
    int e = blockIdx.x * blockDim.x + threadIdx.x;
    if (e < nE) unsafeAtomicAdd(&deg[dst[e]], 1.0f);
}

// one wave per (day, edge): add Y[d, src[e], :] into Agg[d, dst[e], :]
__global__ void k_scatter(const float* __restrict__ Y, const int* __restrict__ src,
                          const int* __restrict__ dst, float* __restrict__ Agg, int nE) {
    int task = blockIdx.x * (blockDim.x >> 5) + (threadIdx.x >> 5);
    if (task >= nE * NUM_DAYS) return;
    int e = task % nE;
    int d = task / nE;
    int lane = threadIdx.x & 31;
    const float* y = Y   + ((size_t)d * N_STOCKS + src[e]) * HID;
    float*       a = Agg + ((size_t)d * N_STOCKS + dst[e]) * HID;
#pragma unroll
    for (int h = lane; h < HID; h += 32)
        unsafeAtomicAdd(&a[h], y[h]);
}

// Sb = bf16( Y + Agg / max(deg,1) )
__global__ void k_combine(const float* __restrict__ Y, const float* __restrict__ Agg,
                          const float* __restrict__ deg, bf16* __restrict__ Sb, int total) {
    int i = blockIdx.x * blockDim.x + threadIdx.x;
    if (i >= total) return;
    int n = (i >> 9) & (N_STOCKS - 1);
    float dg = fmaxf(deg[n], 1.0f);
    Sb[i] = (bf16)(Y[i] + Agg[i] / dg);
}

// day-mix 1x3 conv over hidden dim (pad 1), then lrelu -> bf16
__global__ void k_conv(const float* __restrict__ Sp, const float* __restrict__ Wc,
                       const float* __restrict__ bc, bf16* __restrict__ Cb) {
    int i = blockIdx.x * blockDim.x + threadIdx.x;   // n*512 + h
    if (i >= N_STOCKS * HID) return;
    int h = i & (HID - 1), n = i >> 9;
    float acc = bc[0];
#pragma unroll
    for (int d = 0; d < NUM_DAYS; ++d) {
        const float* row = Sp + ((size_t)d * N_STOCKS + n) * HID;
        float xm = (h > 0)       ? row[h - 1] : 0.f;
        float x0 =                 row[h];
        float xp = (h < HID - 1) ? row[h + 1] : 0.f;
        acc += Wc[d * 3] * xm + Wc[d * 3 + 1] * x0 + Wc[d * 3 + 2] * xp;
    }
    Cb[i] = (bf16)lrelu(acc);
}

// final GEMV: out[n] = sum_k O[n,k] * W4[k] + b4
__global__ void k_final(const float* __restrict__ O, const float* __restrict__ W4,
                        const float* __restrict__ b4, float* __restrict__ out) {
    int n = blockIdx.x;
    int lane = threadIdx.x;
    const float* row = O + (size_t)n * HID2;
    float s = 0.f;
#pragma unroll
    for (int k = lane; k < HID2; k += 32) s += row[k] * W4[k];
#pragma unroll
    for (int off = 16; off > 0; off >>= 1) s += __shfl_down(s, off, 32);
    if (lane == 0) out[n] = s + b4[0];
}

// ---------------- host launcher ----------------

extern "C" void kernel_launch(void* const* d_in, const int* in_sizes, int n_in,
                              void* d_out, int out_size, void* d_ws, size_t ws_size,
                              hipStream_t stream) {
    (void)in_sizes; (void)n_in; (void)out_size; (void)ws_size;

    const float* x  = (const float*)d_in[0];
    const int*   src= (const int*)d_in[1];
    const int*   dst= (const int*)d_in[2];
    const float* W1 = (const float*)d_in[3];
    const float* b1 = (const float*)d_in[4];
    const float* W2 = (const float*)d_in[5];
    const float* b2 = (const float*)d_in[6];
    const float* Wg = (const float*)d_in[7];
    const float* bg = (const float*)d_in[8];
    const float* Wc = (const float*)d_in[9];
    const float* bc = (const float*)d_in[10];
    const float* W3 = (const float*)d_in[11];
    const float* b3 = (const float*)d_in[12];
    const float* W4 = (const float*)d_in[13];
    const float* b4 = (const float*)d_in[14];
    float* out = (float*)d_out;

    char* ws = (char*)d_ws;
    size_t off = 0;
    auto alloc = [&](size_t bytes) -> char* {
        char* p = ws + off;
        off += (bytes + 255) & ~(size_t)255;
        return p;
    };

    bf16*  W1b = (bf16*)alloc((size_t)HID * IN_DIM * 2);
    bf16*  W2b = (bf16*)alloc((size_t)HID * HID * 2);
    bf16*  Wgb = (bf16*)alloc((size_t)HID * HID * 2);
    bf16*  W3b = (bf16*)alloc((size_t)HID2 * HID * 2);
    bf16*  Xb  = (bf16*)alloc((size_t)MROWS * IN_DIM * 2);
    bf16*  H1b = (bf16*)alloc((size_t)MROWS * HID * 2);
    float* Yf  = (float*)alloc((size_t)MROWS * HID * 4);
    float* Agg = (float*)alloc((size_t)MROWS * HID * 4);
    float* deg = (float*)alloc((size_t)N_STOCKS * 4);
    bf16*  Cb  = (bf16*)alloc((size_t)N_STOCKS * HID * 2);
    float* Of  = (float*)alloc((size_t)N_STOCKS * HID2 * 4);
    bf16*  Sb  = H1b;   // alias: H1 dead after GEMM2
    float* Sp  = Agg;   // alias: Agg dead after combine

    auto cvt = [&](const float* in, bf16* o, int n) {
        k_f32_to_bf16<<<(n + 255) / 256, 256, 0, stream>>>(in, o, n);
    };
    cvt(W1, W1b, HID * IN_DIM);
    cvt(W2, W2b, HID * HID);
    cvt(Wg, Wgb, HID * HID);
    cvt(W3, W3b, HID2 * HID);
    cvt(x,  Xb,  MROWS * IN_DIM);

    // h1 = lrelu(x W1^T + b1)
    k_gemm_wmma<true, true ><<<dim3(HID / 256, MROWS / 16), 128, 0, stream>>>(
        Xb, W1b, b1, H1b, MROWS, HID, IN_DIM);
    // y = lrelu(h1 W2^T + b2) (kept f32 for scatter accumulation)
    k_gemm_wmma<true, false><<<dim3(HID / 256, MROWS / 16), 128, 0, stream>>>(
        H1b, W2b, b2, Yf, MROWS, HID, HID);

    // graph aggregation
    k_fill_f32<<<(N_STOCKS + 255) / 256, 256, 0, stream>>>(deg, 0.f, N_STOCKS);
    k_fill_f32<<<(MROWS * HID + 255) / 256, 256, 0, stream>>>(Agg, 0.f, MROWS * HID);
    k_deg<<<(N_EDGES + 255) / 256, 256, 0, stream>>>(dst, deg, N_EDGES);
    {
        int tasks = N_EDGES * NUM_DAYS;          // one wave per (day, edge)
        int blocks = (tasks + 7) / 8;            // 8 waves / 256-thread block
        k_scatter<<<blocks, 256, 0, stream>>>(Yf, src, dst, Agg, N_EDGES);
    }
    k_combine<<<(MROWS * HID + 255) / 256, 256, 0, stream>>>(Yf, Agg, deg, Sb, MROWS * HID);

    // spatial = lrelu((y+agg) Wg^T + bg)
    k_gemm_wmma<true, false><<<dim3(HID / 256, MROWS / 16), 128, 0, stream>>>(
        Sb, Wgb, bg, Sp, MROWS, HID, HID);

    // day-conv + lrelu
    k_conv<<<(N_STOCKS * HID + 255) / 256, 256, 0, stream>>>(Sp, Wc, bc, Cb);

    // lrelu(c W3^T + b3)
    k_gemm_wmma<true, false><<<dim3(HID2 / 256, N_STOCKS / 16), 128, 0, stream>>>(
        Cb, W3b, b3, Of, N_STOCKS, HID2, HID);

    // final linear
    k_final<<<N_STOCKS, 32, 0, stream>>>(Of, W4, b4, out);
}
